// FastAttention_67611375173885
// MI455X (gfx1250) — matrix operand order repaired and checked by reading, not compiled, not run on hardware
//
#include <hip/hip_runtime.h>
#include <hip/hip_bf16.h>

// Performer (FAVOR+) fast attention for MI455X (gfx1250, wave32, WMMA).
// B=2, N=8192, H=8, D=64, M=256. All GEMMs via v_wmma_f32_16x16x32_bf16.

#define BB 2
#define HH 8
#define BHN 16
#define NN 8192
#define DD 64
#define MM 256
#define RN 128              // rows (n) per block tile
#define NBLK (NN / RN)      // 64

#define C_NORM 0.35355339059327379f   // D^-0.25
#define DIAG_C 0.0625f                // 0.5 * D^-0.5
#define RATIO  0.0625f                // M^-0.5
#define EPSV   1e-4f

// padded LDS leading dims (elements). A-operand needs ld%8==0, B-operand ld%16==0.
#define PROJ_LD 80     // proj  [256][64]  (B operand)
#define TILE_LD 72     // q/k   [128][64]  (A operand)
#define KFT_LD  144    // kf^T  [256][128] (B operand)
#define VT_LD   144    // v^T   [64][128]  (A operand)
#define QF_LD   264    // qf    [128][256] (A operand)
#define CTX_LD  272    // ctx^T [64][256]  (B operand)

typedef __attribute__((ext_vector_type(16))) __bf16 v16bf;
typedef __attribute__((ext_vector_type(8)))  __bf16 v8bf;
typedef __attribute__((ext_vector_type(8)))  float  v8f;

__device__ __forceinline__ __bf16 to_bf16(float f) {
    union { float f; unsigned u; } a; a.f = f;
    unsigned r = a.u + 0x7FFFu + ((a.u >> 16) & 1u);   // RNE
    union { unsigned short s; __bf16 b; } o; o.s = (unsigned short)(r >> 16);
    return o.b;
}
__device__ __forceinline__ float bf16_to_f(__bf16 x) {
    union { unsigned short s; __bf16 b; } i; i.b = x;
    union { unsigned u; float f; } o; o.u = ((unsigned)i.s) << 16;
    return o.f;
}
// order-preserving float<->uint map for atomicMax on signed floats
__device__ __forceinline__ unsigned f2ord(float f) {
    unsigned u = __float_as_uint(f);
    return u ^ ((u >> 31) ? 0xFFFFFFFFu : 0x80000000u);
}
__device__ __forceinline__ float ord2f(unsigned u) {
    unsigned b = (u & 0x80000000u) ? (u ^ 0x80000000u) : ~u;
    return __uint_as_float(b);
}

// A fragment (16x32 bf16) from row-major [16+][K] LDS matrix at (row0,col0)=p.
// lane L: row = L&15; chunks of 8 at cols (L>>4)*8 and 16+(L>>4)*8.
__device__ __forceinline__ v16bf loadA(const __bf16* p, int ld, int lane) {
    int r = lane & 15, s = lane >> 4;
    const __bf16* row = p + r * ld + s * 8;
    v8bf c0 = *(const v8bf*)(row);
    v8bf c1 = *(const v8bf*)(row + 16);
    return __builtin_shufflevector(c0, c1, 0,1,2,3,4,5,6,7,8,9,10,11,12,13,14,15);
}
// B fragment (32x16 bf16) == X^T for row-major X [16+][K] at (row0,col0)=p.
// lane L: X row (L&15); chunks of 8 at cols (L>>4)*8 and 16+(L>>4)*8.
__device__ __forceinline__ v16bf loadB(const __bf16* p, int ld, int lane) {
    int r = lane & 15, s = lane >> 4;
    const __bf16* row = p + r * ld + s * 8;
    v8bf c0 = *(const v8bf*)(row);
    v8bf c1 = *(const v8bf*)(row + 16);
    return __builtin_shufflevector(c0, c1, 0,1,2,3,4,5,6,7,8,9,10,11,12,13,14,15);
}

__device__ __forceinline__ v8f wmma_bf16(v16bf a, v16bf b, v8f c) {
    return __builtin_amdgcn_wmma_f32_16x16x32_bf16(false, a, false, b, (short)0, c,
                                                   false, false);
}

// ---------------------------------------------------------------- init
__global__ __launch_bounds__(256) void perf_init(unsigned* __restrict__ amaxu,
                                                 float* __restrict__ ksum,
                                                 float* __restrict__ gctx) {
    int i = blockIdx.x * 256 + threadIdx.x;
    if (i < BHN) amaxu[i] = f2ord(-3.0e38f);
    if (i < BHN * MM) ksum[i] = 0.0f;
    if (i < BHN * DD * MM) gctx[i] = 0.0f;
}

// ---------------------------------------------------------------- pass 1: k amax
__global__ __launch_bounds__(256) void perf_kmax(const float* __restrict__ kin,
                                                 const float* __restrict__ proj,
                                                 unsigned* __restrict__ amaxu) {
    __shared__ __align__(32) __bf16 s_proj[MM * PROJ_LD];
    __shared__ __align__(32) __bf16 s_tile[RN * TILE_LD];
    __shared__ float s_wmax[8];

    int tid = threadIdx.x, lane = tid & 31, w = tid >> 5;
    int bh = blockIdx.y, b = bh >> 3, h = bh & 7;
    int n0 = blockIdx.x * RN;

    for (int i = tid; i < MM * DD; i += 256) {
        int m = i >> 6, d = i & 63;
        s_proj[m * PROJ_LD + d] = to_bf16(proj[i]);
    }
    for (int i = tid; i < RN * DD; i += 256) {
        int r = i >> 6, c = i & 63;
        float g = kin[(((size_t)b * NN + (n0 + r)) * HH + h) * DD + c];
        s_tile[r * TILE_LD + c] = to_bf16(g);
    }
    __syncthreads();

    float lmax = -3.0e38f;
    for (int t = 0; t < 16; ++t) {
        v8f acc = {0.f, 0.f, 0.f, 0.f, 0.f, 0.f, 0.f, 0.f};
        for (int kk = 0; kk < DD; kk += 32) {
            v16bf a = loadA(&s_tile[(w * 16) * TILE_LD + kk], TILE_LD, lane);
            v16bf bm = loadB(&s_proj[(t * 16) * PROJ_LD + kk], PROJ_LD, lane);
            acc = wmma_bf16(a, bm, acc);
        }
        for (int r = 0; r < 8; ++r) lmax = fmaxf(lmax, acc[r]);
    }
    lmax *= C_NORM;  // positive scale commutes with max
    for (int off = 16; off >= 1; off >>= 1)
        lmax = fmaxf(lmax, __shfl_xor(lmax, off, 32));
    if (lane == 0) s_wmax[w] = lmax;
    __syncthreads();
    if (tid == 0) {
        float m = s_wmax[0];
        for (int i = 1; i < 8; ++i) m = fmaxf(m, s_wmax[i]);
        atomicMax(amaxu + bh, f2ord(m));
    }
}

// ---------------------------------------------------------------- pass 2: kf, ksum, context
__global__ __launch_bounds__(256) void perf_kctx(const float* __restrict__ kin,
                                                 const float* __restrict__ vin,
                                                 const float* __restrict__ proj,
                                                 const unsigned* __restrict__ amaxu,
                                                 float* __restrict__ ksum,
                                                 float* __restrict__ gctx) {
    __shared__ __align__(32) __bf16 s_proj[MM * PROJ_LD];
    __shared__ __align__(32) __bf16 s_tile[RN * TILE_LD];
    __shared__ __align__(32) __bf16 s_kfT[MM * KFT_LD];
    __shared__ __align__(32) __bf16 s_vT[DD * VT_LD];
    __shared__ float s_diag[RN];

    int tid = threadIdx.x, lane = tid & 31, w = tid >> 5;
    int j = lane & 15, hi = lane >> 4;
    int bh = blockIdx.y, b = bh >> 3, h = bh & 7;
    int n0 = blockIdx.x * RN;

    if (tid < RN) s_diag[tid] = 0.0f;
    __syncthreads();

    for (int i = tid; i < MM * DD; i += 256) {
        int m = i >> 6, d = i & 63;
        s_proj[m * PROJ_LD + d] = to_bf16(proj[i]);
    }
    for (int i = tid; i < RN * DD; i += 256) {
        int r = i >> 6, c = i & 63;
        float g = kin[(((size_t)b * NN + (n0 + r)) * HH + h) * DD + c];
        s_tile[r * TILE_LD + c] = to_bf16(g);
        atomicAdd(&s_diag[r], g * g);
    }
    for (int i = tid; i < RN * DD; i += 256) {
        int r = i >> 6, c = i & 63;
        float g = vin[(((size_t)b * NN + (n0 + r)) * HH + h) * DD + c];
        s_vT[c * VT_LD + r] = to_bf16(g);   // transpose stage
    }
    __syncthreads();

    float amax = ord2f(amaxu[bh]);

    v8f dash[16];
    for (int t = 0; t < 16; ++t) {
        v8f acc = {0.f, 0.f, 0.f, 0.f, 0.f, 0.f, 0.f, 0.f};
        for (int kk = 0; kk < DD; kk += 32) {
            v16bf a = loadA(&s_tile[(w * 16) * TILE_LD + kk], TILE_LD, lane);
            v16bf bm = loadB(&s_proj[(t * 16) * PROJ_LD + kk], PROJ_LD, lane);
            acc = wmma_bf16(a, bm, acc);
        }
        dash[t] = acc;
    }
    // kf = ratio*(exp(c*dash - diag - amax) + eps)  -> kf^T in LDS
    for (int t = 0; t < 16; ++t) {
        for (int r = 0; r < 8; ++r) {
            int row = w * 16 + r + 8 * hi;
            float dg = DIAG_C * s_diag[row];
            float e = RATIO * (__expf(C_NORM * dash[t][r] - dg - amax) + EPSV);
            s_kfT[(t * 16 + j) * KFT_LD + row] = to_bf16(e);
        }
    }
    __syncthreads();

    // k_cumsum partial: one thread per m
    {
        float ss = 0.0f;
        const __bf16* rowp = &s_kfT[tid * KFT_LD];
        for (int r = 0; r < RN; ++r) ss += bf16_to_f(rowp[r]);
        atomicAdd(&ksum[bh * MM + tid], ss);
    }

    // context^T partial: ctxT[d][m] += sum_n vT[d][n]*kfT[m][n]
    for (int T = w; T < 64; T += 8) {
        int dt = T >> 4, mt = T & 15;
        v8f acc = {0.f, 0.f, 0.f, 0.f, 0.f, 0.f, 0.f, 0.f};
        for (int kk = 0; kk < RN; kk += 32) {
            v16bf a = loadA(&s_vT[(dt * 16) * VT_LD + kk], VT_LD, lane);
            v16bf bm = loadB(&s_kfT[(mt * 16) * KFT_LD + kk], KFT_LD, lane);
            acc = wmma_bf16(a, bm, acc);
        }
        for (int r = 0; r < 8; ++r) {
            int d = dt * 16 + r + 8 * hi;
            int m = mt * 16 + j;
            atomicAdd(&gctx[((size_t)bh * DD + d) * MM + m], acc[r]);
        }
    }
}

// ---------------------------------------------------------------- pass 3: qf, denom, output
__global__ __launch_bounds__(256) void perf_qout(const float* __restrict__ qin,
                                                 const float* __restrict__ proj,
                                                 const float* __restrict__ ksum,
                                                 const float* __restrict__ gctx,
                                                 float* __restrict__ out) {
    __shared__ __align__(32) __bf16 s_proj[MM * PROJ_LD];
    __shared__ __align__(32) __bf16 s_tile[RN * TILE_LD];
    __shared__ __align__(32) __bf16 s_qf[RN * QF_LD];
    __shared__ __align__(32) __bf16 s_ctxT[DD * CTX_LD];
    __shared__ float s_diag[RN];

    int tid = threadIdx.x, lane = tid & 31, w = tid >> 5;
    int j = lane & 15, hi = lane >> 4;
    int bh = blockIdx.y, b = bh >> 3, h = bh & 7;
    int n0 = blockIdx.x * RN;

    if (tid < RN) s_diag[tid] = 0.0f;
    __syncthreads();

    for (int i = tid; i < MM * DD; i += 256) {
        int m = i >> 6, d = i & 63;
        s_proj[m * PROJ_LD + d] = to_bf16(proj[i]);
    }
    for (int i = tid; i < RN * DD; i += 256) {
        int r = i >> 6, c = i & 63;
        float g = qin[(((size_t)b * NN + (n0 + r)) * HH + h) * DD + c];
        s_tile[r * TILE_LD + c] = to_bf16(g);
        atomicAdd(&s_diag[r], g * g);
    }
    for (int i = tid; i < DD * MM; i += 256) {
        int d = i >> 8, m = i & 255;
        s_ctxT[d * CTX_LD + m] = to_bf16(gctx[(size_t)bh * DD * MM + i]);
    }
    __syncthreads();

    v8f dash[16];
    for (int t = 0; t < 16; ++t) {
        v8f acc = {0.f, 0.f, 0.f, 0.f, 0.f, 0.f, 0.f, 0.f};
        for (int kk = 0; kk < DD; kk += 32) {
            v16bf a = loadA(&s_tile[(w * 16) * TILE_LD + kk], TILE_LD, lane);
            v16bf bm = loadB(&s_proj[(t * 16) * PROJ_LD + kk], PROJ_LD, lane);
            acc = wmma_bf16(a, bm, acc);
        }
        dash[t] = acc;
    }

    // per-row max over M (rows live in the C-fragment as r + 8*hi; cols as lanes&15)
    float rmax[8], rsum[8], dinv[8];
    for (int r = 0; r < 8; ++r) {
        float vmax = -3.0e38f;
        for (int t = 0; t < 16; ++t) vmax = fmaxf(vmax, dash[t][r]);
        for (int off = 1; off < 16; off <<= 1)
            vmax = fmaxf(vmax, __shfl_xor(vmax, off, 32));  // stays within 16-lane half
        rmax[r] = C_NORM * vmax;
        rsum[r] = 0.0f;
    }

    // qf + denominator accumulation
    for (int t = 0; t < 16; ++t) {
        float ks = ksum[bh * MM + t * 16 + j];
        for (int r = 0; r < 8; ++r) {
            int row = w * 16 + r + 8 * hi;
            float dg = DIAG_C * s_diag[row];
            float e = RATIO * (__expf(C_NORM * dash[t][r] - dg - rmax[r]) + EPSV);
            rsum[r] += e * ks;
            s_qf[row * QF_LD + t * 16 + j] = to_bf16(e);
        }
    }
    for (int r = 0; r < 8; ++r) {
        float ss = rsum[r];
        for (int off = 1; off < 16; off <<= 1) ss += __shfl_xor(ss, off, 32);
        dinv[r] = 1.0f / ss;
    }
    __syncthreads();

    // out[n][d] = dinv[n] * sum_m qf[n][m] * ctxT[d][m]
    for (int dt = 0; dt < 4; ++dt) {
        v8f acc = {0.f, 0.f, 0.f, 0.f, 0.f, 0.f, 0.f, 0.f};
        for (int kk = 0; kk < MM; kk += 32) {
            v16bf a = loadA(&s_qf[(w * 16) * QF_LD + kk], QF_LD, lane);
            v16bf bm = loadB(&s_ctxT[(dt * 16) * CTX_LD + kk], CTX_LD, lane);
            acc = wmma_bf16(a, bm, acc);
        }
        for (int r = 0; r < 8; ++r) {
            int row = w * 16 + r + 8 * hi;
            int n = n0 + row;
            int d = dt * 16 + j;
            out[(((size_t)b * NN + n) * HH + h) * DD + d] = acc[r] * dinv[r];
        }
    }
}

// ---------------------------------------------------------------- launch
extern "C" void kernel_launch(void* const* d_in, const int* in_sizes, int n_in,
                              void* d_out, int out_size, void* d_ws, size_t ws_size,
                              hipStream_t stream) {
    const float* q    = (const float*)d_in[0];
    const float* k    = (const float*)d_in[1];
    const float* v    = (const float*)d_in[2];
    const float* proj = (const float*)d_in[3];
    float* out = (float*)d_out;

    // workspace: amax (order-mapped uints) | k_cumsum | context^T  (~1.02 MB)
    unsigned* amaxu = (unsigned*)d_ws;
    float* ksum = (float*)((char*)d_ws + 256);
    float* gctx = (float*)((char*)d_ws + 256 + (size_t)BHN * MM * sizeof(float));

    perf_init<<<(BHN * DD * MM + 255) / 256, 256, 0, stream>>>(amaxu, ksum, gctx);
    dim3 grid(NBLK, BHN);
    perf_kmax<<<grid, 256, 0, stream>>>(k, proj, amaxu);
    perf_kctx<<<grid, 256, 0, stream>>>(k, v, proj, amaxu, ksum, gctx);
    perf_qout<<<grid, 256, 0, stream>>>(q, proj, ksum, gctx, out);
}
